// RRM_14276471292102
// MI455X (gfx1250) — matrix operand order repaired and checked by reading, not compiled, not run on hardware
//
#include <hip/hip_runtime.h>
#include <hip/hip_bf16.h>

// ---------------- problem constants ----------------
constexpr int cB  = 8;
constexpr int cC  = 64;
constexpr int cH  = 150;
constexpr int cW  = 150;
constexpr int cHW = cH * cW;          // 22500
constexpr int cM  = cB * cHW;         // 180000 rows, divisible by 16
constexpr float cEPS = 1e-5f;

typedef __bf16 bf16_t;
typedef unsigned int u32;
typedef __attribute__((ext_vector_type(16))) __bf16 v16bf;
typedef __attribute__((ext_vector_type(8)))  float  v8f;
typedef __attribute__((ext_vector_type(8)))  u32    v8u;
typedef __attribute__((ext_vector_type(4)))  float  f4;

__device__ __forceinline__ float sigm(float x) { return 1.f / (1.f + __expf(-x)); }

__device__ __forceinline__ u32 pack_bf16(float a, float b) {
    u32 lo = (u32)__builtin_bit_cast(unsigned short, (bf16_t)a);
    u32 hi = (u32)__builtin_bit_cast(unsigned short, (bf16_t)b);
    return lo | (hi << 16);
}

// ---------------- utility kernels ----------------
__global__ void k_zero(float* p, int n) {
    int i = blockIdx.x * blockDim.x + threadIdx.x;
    if (i < n) p[i] = 0.f;
}

// NCHW (C=64) -> channels-last [M, ldo] at column offset
__global__ void k_pack_nchw(const float* __restrict__ src, float* __restrict__ dst,
                            int ldo, int coff) {
    int i = blockIdx.x * blockDim.x + threadIdx.x;
    if (i >= cM * cC) return;
    int m = i >> 6, c = i & 63;
    int b = m / cHW, hw = m - b * cHW;
    dst[(size_t)m * ldo + coff + c] = src[(size_t)(b * cC + c) * cHW + hw];
}

// OIHW weights -> packed-bf16 B operand [Kpad/2, Npad] dwords,
// kk = (kh*KW+kw)*I + ci ; dword(kp,n) = bf16(B[2kp][n]) | bf16(B[2kp+1][n])<<16
__global__ void k_repack_pk(const float* __restrict__ w, u32* __restrict__ Bpk,
                            int O, int I, int KH, int KW, int Kpad, int Npad) {
    int i = blockIdx.x * blockDim.x + threadIdx.x;
    if (i >= (Kpad >> 1) * Npad) return;
    int n = i % Npad, kp = i / Npad;
    int Kd = I * KH * KW;
    float v[2] = {0.f, 0.f};
    if (n < O) {
#pragma unroll
        for (int t = 0; t < 2; ++t) {
            int kk = 2 * kp + t;
            if (kk < Kd) {
                int tap = kk / I, ci = kk - tap * I;
                int kh = tap / KW, kw = tap - kh * KW;
                v[t] = w[(((size_t)n * I + ci) * KH + kh) * KW + kw];
            }
        }
    }
    Bpk[i] = pack_bf16(v[0], v[1]);
}

// row-major [Kd,N] matrix -> packed-bf16 B operand [Kpad/2, Npad]
__global__ void k_pack_bmat(const float* __restrict__ W, u32* __restrict__ Bpk,
                            int Kd, int N, int Kpad, int Npad) {
    int i = blockIdx.x * blockDim.x + threadIdx.x;
    if (i >= (Kpad >> 1) * Npad) return;
    int n = i % Npad, kp = i / Npad;
    float v0 = (2 * kp < Kd && n < N) ? W[(size_t)(2 * kp) * N + n] : 0.f;
    float v1 = (2 * kp + 1 < Kd && n < N) ? W[(size_t)(2 * kp + 1) * N + n] : 0.f;
    Bpk[i] = pack_bf16(v0, v1);
}

// ---------------- WMMA GEMM (all dims compile-time): C[M,LDC] (+)= A x B ----------------
template <int LDA, int LDB, int LDC, int KK>
__global__ void k_gemm(const float* __restrict__ A, const u32* __restrict__ Bpk,
                       float* __restrict__ Cm, int Ntiles, int nwaves, int accumulate) {
    int wave = (int)((blockIdx.x * blockDim.x + threadIdx.x) >> 5);
    if (wave >= nwaves) return;
    int lane = threadIdx.x & 31;
    int mt = wave / Ntiles, nt = wave - mt * Ntiles;
    int m0 = mt << 4, n0 = nt << 4;
    int half = lane >> 4, l15 = lane & 15;
    int ncol = n0 + l15;
    const float* Abase = A + (size_t)(m0 + l15) * LDA + half * 8;
    float* Cbase = Cm + (size_t)(m0 + 8 * half) * LDC + ncol;
    v8f acc;
    if (accumulate) {
#pragma unroll
        for (int r = 0; r < 8; ++r) acc[r] = Cbase[r * LDC];
    } else {
#pragma unroll
        for (int r = 0; r < 8; ++r) acc[r] = 0.f;
    }
    const u32* Bbase = Bpk + (size_t)(4 * half) * LDB + ncol;
#pragma unroll
    for (int k0 = 0; k0 < KK; k0 += 32) {
        // A: two contiguous 8-float runs per half-wave -> 4x b128 loads
        const f4* Ar = (const f4*)(Abase + k0);
        f4 a0 = Ar[0], a1 = Ar[1];
        const f4* Ar2 = (const f4*)(Abase + k0 + 16);
        f4 a2 = Ar2[0], a3 = Ar2[1];
        v16bf av;
#pragma unroll
        for (int t = 0; t < 4; ++t) {
            av[t]      = (bf16_t)a0[t];
            av[4 + t]  = (bf16_t)a1[t];
            av[8 + t]  = (bf16_t)a2[t];
            av[12 + t] = (bf16_t)a3[t];
        }
        // B: pre-packed bf16 pairs, one dword per VGPR, immediate offsets
        const u32* Br = Bbase + (size_t)(k0 >> 1) * LDB;
        v8u bu;
#pragma unroll
        for (int v = 0; v < 8; ++v) bu[v] = Br[(v + (v & 4)) * LDB];
        v16bf bv = __builtin_bit_cast(v16bf, bu);
        acc = __builtin_amdgcn_wmma_f32_16x16x32_bf16(false, av, false, bv,
                                                      (short)0, acc, false, false);
    }
#pragma unroll
    for (int r = 0; r < 8; ++r) Cbase[r * LDC] = acc[r];
}

// ---------------- WMMA implicit-GEMM 3x3 conv (pad=1), channels-last in [M,LDIN]
template <int LDIN, int CIN, int KPAD, int LDB, int LDC>
__global__ void k_conv3(const float* __restrict__ In, const u32* __restrict__ Bpk,
                        float* __restrict__ Cm, int Ntiles, int nwaves) {
    constexpr int Kd = 9 * CIN;
    int wave = (int)((blockIdx.x * blockDim.x + threadIdx.x) >> 5);
    if (wave >= nwaves) return;
    int lane = threadIdx.x & 31;
    int mt = wave / Ntiles, nt = wave - mt * Ntiles;
    int m0 = mt << 4, n0 = nt << 4;
    int half = lane >> 4, l15 = lane & 15;
    int ncol = n0 + l15;
    int mrow = m0 + l15;
    int b = mrow / cHW, hw = mrow - b * cHW;
    int h = hw / cW, w = hw - h * cW;
    // K-pairs share a tap (CIN even) and the same spatial gate -> b64 gathers
    auto fetch2 = [&](int kk) -> float2 {
        float2 z; z.x = 0.f; z.y = 0.f;
        if (kk >= Kd) return z;
        int tap = kk / CIN, ci = kk - tap * CIN;
        int dh = tap / 3 - 1, dw = tap - (tap / 3) * 3 - 1;
        int hh = h + dh, ww = w + dw;
        if (hh < 0 || hh >= cH || ww < 0 || ww >= cW) return z;
        return *(const float2*)(In + (size_t)(mrow + dh * cW + dw) * LDIN + ci);
    };
    const u32* Bbase = Bpk + (size_t)(4 * half) * LDB + ncol;
    v8f acc;
#pragma unroll
    for (int r = 0; r < 8; ++r) acc[r] = 0.f;
    for (int k0 = 0; k0 < KPAD; k0 += 32) {
        v16bf av;
#pragma unroll
        for (int v = 0; v < 4; ++v) {
            float2 p = fetch2(k0 + half * 8 + 2 * v);
            av[2 * v]     = (bf16_t)p.x;
            av[2 * v + 1] = (bf16_t)p.y;
            float2 q = fetch2(k0 + 16 + half * 8 + 2 * v);
            av[8 + 2 * v]     = (bf16_t)q.x;
            av[8 + 2 * v + 1] = (bf16_t)q.y;
        }
        const u32* Br = Bbase + (size_t)(k0 >> 1) * LDB;
        v8u bu;
#pragma unroll
        for (int v = 0; v < 8; ++v) bu[v] = Br[(v + (v & 4)) * LDB];
        v16bf bv = __builtin_bit_cast(v16bf, bu);
        acc = __builtin_amdgcn_wmma_f32_16x16x32_bf16(false, av, false, bv,
                                                      (short)0, acc, false, false);
    }
    float* Cbase = Cm + (size_t)(m0 + 8 * half) * LDC + ncol;
#pragma unroll
    for (int r = 0; r < 8; ++r) Cbase[r * LDC] = acc[r];
}

// ---------------- batch-norm statistics per column, then fused norm+activation
__global__ void k_colstats(const float* __restrict__ X, int ld,
                           const float* __restrict__ g, const float* __restrict__ b,
                           float* __restrict__ scale, float* __restrict__ shift) {
    __shared__ float s1[256];
    __shared__ float s2[256];
    int n = blockIdx.x, tid = threadIdx.x;
    float sum = 0.f, sq = 0.f;
    for (int m = tid; m < cM; m += 256) {
        float v = X[(size_t)m * ld + n];
        sum += v; sq += v * v;
    }
    s1[tid] = sum; s2[tid] = sq;
    __syncthreads();
    for (int st = 128; st > 0; st >>= 1) {
        if (tid < st) { s1[tid] += s1[tid + st]; s2[tid] += s2[tid + st]; }
        __syncthreads();
    }
    if (tid == 0) {
        float mean = s1[0] / (float)cM;
        float var  = s2[0] / (float)cM - mean * mean;
        float sc = g[n] * rsqrtf(var + cEPS);
        scale[n] = sc;
        shift[n] = b[n] - mean * sc;
    }
}

// mode 0: relu; mode 1: sigmoid(relu(x)); optional per-row scale
__global__ void k_bn_act(const float* __restrict__ X, int ldx,
                         float* __restrict__ Y, int ldy, int N,
                         const float* __restrict__ scale, const float* __restrict__ shift,
                         int mode, const float* __restrict__ rows) {
    int i = blockIdx.x * blockDim.x + threadIdx.x;
    if (i >= cM * N) return;
    int m = i / N, c = i - m * N;
    float v = X[(size_t)m * ldx + c] * scale[c] + shift[c];
    v = fmaxf(v, 0.f);
    if (mode == 1) v = sigm(v);
    if (rows) v *= rows[m];
    Y[(size_t)m * ldy + c] = v;
}

// ---------------- CCM helpers ----------------
__global__ void k_colpool(const float* __restrict__ X, float* __restrict__ avg,
                          float* __restrict__ mxo) {
    __shared__ float s1[256];
    __shared__ float s2[256];
    int bc = blockIdx.x;            // b*64+c
    int b = bc >> 6, c = bc & 63, tid = threadIdx.x;
    float sum = 0.f, mx = -1e30f;
    for (int t = tid; t < cHW; t += 256) {
        float v = X[(size_t)(b * cHW + t) * 64 + c];
        sum += v; mx = fmaxf(mx, v);
    }
    s1[tid] = sum; s2[tid] = mx;
    __syncthreads();
    for (int st = 128; st > 0; st >>= 1) {
        if (tid < st) { s1[tid] += s1[tid + st]; s2[tid] = fmaxf(s2[tid], s2[tid + st]); }
        __syncthreads();
    }
    if (tid == 0) { avg[bc] = s1[0] / (float)cHW; mxo[bc] = s2[0]; }
}

// pooled [B,64,s,s] (mean & max over contiguous (H/s)x(W/s) blocks), input scaled per (b,c)
__global__ void k_blockpool(const float* __restrict__ X, const float* __restrict__ sc,
                            int s, float* __restrict__ avg, float* __restrict__ mxo) {
    __shared__ float s1[256];
    __shared__ float s2[256];
    int id = blockIdx.x;
    int j = id % s, i2 = (id / s) % s;
    int c = (id / (s * s)) & 63, b = id / (s * s * 64);
    int bh = cH / s, bw = cW / s;
    float f = sc ? sc[b * 64 + c] : 1.f;
    int tid = threadIdx.x;
    float sum = 0.f, mx = -1e30f;
    for (int t = tid; t < bh * bw; t += 256) {
        int hh = i2 * bh + t / bw;
        int ww = j * bw + t % bw;
        float v = X[(size_t)(b * cHW + hh * cW + ww) * 64 + c] * f;
        sum += v; mx = fmaxf(mx, v);
    }
    s1[tid] = sum; s2[tid] = mx;
    __syncthreads();
    for (int st = 128; st > 0; st >>= 1) {
        if (tid < st) { s1[tid] += s1[tid + st]; s2[tid] = fmaxf(s2[tid], s2[tid + st]); }
        __syncthreads();
    }
    if (tid == 0) { avg[id] = s1[0] / (float)(bh * bw); mxo[id] = s2[0]; }
}

// tiny NCHW conv 1x1 (single block)
__global__ void k_tiny_conv1x1(const float* __restrict__ in, const float* __restrict__ w,
                               float* __restrict__ out, int Bc, int Cin, int Cout, int S2) {
    int total = Bc * Cout * S2;
    for (int i = threadIdx.x; i < total; i += blockDim.x) {
        int s = i % S2, o = (i / S2) % Cout, b = i / (S2 * Cout);
        float acc = 0.f;
        for (int ci = 0; ci < Cin; ++ci)
            acc += in[(b * Cin + ci) * S2 + s] * w[o * Cin + ci];
        out[i] = acc;
    }
}

// tiny valid conv with kernel size == spatial size -> [B,Cout]
__global__ void k_tiny_convk(const float* __restrict__ in, const float* __restrict__ w,
                             float* __restrict__ out, int Bc, int Cin, int Cout, int S) {
    int total = Bc * Cout, S2 = S * S;
    for (int i = threadIdx.x; i < total; i += blockDim.x) {
        int o = i % Cout, b = i / Cout;
        float acc = 0.f;
        for (int ci = 0; ci < Cin; ++ci)
            for (int t = 0; t < S2; ++t)
                acc += in[(b * Cin + ci) * S2 + t] * w[(o * Cin + ci) * S2 + t];
        out[i] = acc;
    }
}

// tiny in-place BN(batch stats over B*S2 per channel)+ReLU (single block)
__global__ void k_tiny_bnrelu(float* __restrict__ x, int Bc, int Cch, int S2,
                              const float* __restrict__ g, const float* __restrict__ bt) {
    for (int o = threadIdx.x; o < Cch; o += blockDim.x) {
        float sum = 0.f, sq = 0.f;
        int n = Bc * S2;
        for (int b = 0; b < Bc; ++b)
            for (int s = 0; s < S2; ++s) {
                float v = x[(b * Cch + o) * S2 + s];
                sum += v; sq += v * v;
            }
        float mean = sum / n, var = sq / n - mean * mean;
        float sc = g[o] * rsqrtf(var + cEPS), sh = bt[o] - mean * sc;
        for (int b = 0; b < Bc; ++b)
            for (int s = 0; s < S2; ++s) {
                int idx = (b * Cch + o) * S2 + s;
                x[idx] = fmaxf(x[idx] * sc + sh, 0.f);
            }
    }
}

__global__ void k_addsig(const float* __restrict__ a, const float* __restrict__ b,
                         float* __restrict__ w, int n) {
    int i = blockIdx.x * blockDim.x + threadIdx.x;
    if (i < n) w[i] = sigm(a[i] + b[i]);
}

// y[m,c] = x[m,c]*s[b*64+c]
__global__ void k_scale_bc(const float* __restrict__ X, const float* __restrict__ s,
                           float* __restrict__ Y) {
    int i = blockIdx.x * blockDim.x + threadIdx.x;
    if (i >= cM * 64) return;
    int m = i >> 6, c = i & 63;
    int b = m / cHW;
    Y[i] = X[i] * s[b * 64 + c];
}

// ---------------- SCM helpers ----------------
__global__ void k_rowreduce(const float* __restrict__ X, float* __restrict__ out2) {
    int m = blockIdx.x * blockDim.x + threadIdx.x;
    if (m >= cM) return;
    const float* r = X + (size_t)m * 64;
    float s = 0.f, mx = -1e30f;
    for (int c = 0; c < 64; ++c) { float v = r[c]; s += v; mx = fmaxf(mx, v); }
    out2[(size_t)m * 2]     = s * (1.f / 64.f);
    out2[(size_t)m * 2 + 1] = mx;
}

__global__ void k_mul2(const float* __restrict__ x2, const float* __restrict__ w,
                       float* __restrict__ y) {
    int i = blockIdx.x * blockDim.x + threadIdx.x;
    if (i >= cM * 2) return;
    y[i] = x2[i] * w[i >> 1];
}

// 2-in 1-out kxk conv, pad=k/2, channels-last [M,2]
__global__ void k_conv2to1(const float* __restrict__ x2, const float* __restrict__ w,
                           int k, float* __restrict__ out) {
    int m = blockIdx.x * blockDim.x + threadIdx.x;
    if (m >= cM) return;
    int hw = m % cHW;
    int h = hw / cW, ww = hw - h * cW;
    int p = k >> 1;
    float acc = 0.f;
    for (int ci = 0; ci < 2; ++ci)
        for (int kh = 0; kh < k; ++kh) {
            int hh = h + kh - p;
            if (hh < 0 || hh >= cH) continue;
            for (int kw = 0; kw < k; ++kw) {
                int wy = ww + kw - p;
                if (wy < 0 || wy >= cW) continue;
                acc += x2[(size_t)(m + (kh - p) * cW + (kw - p)) * 2 + ci] *
                       w[(ci * k + kh) * k + kw];
            }
        }
    out[m] = acc;
}

__global__ void k_make_s12(const float* __restrict__ X, const float* __restrict__ w,
                           float* __restrict__ Y1, float* __restrict__ Y2) {
    int i = blockIdx.x * blockDim.x + threadIdx.x;
    if (i >= cM * 64) return;
    float ws = w[i >> 6];
    float v = X[i];
    Y1[i] = v * ws;
    Y2[i] = v * (1.f - ws);
}

// ---------------- GAT helpers ----------------
__global__ void k_deg(const int* __restrict__ src, float* __restrict__ deg, int E) {
    int e = blockIdx.x * blockDim.x + threadIdx.x;
    if (e < E) atomicAdd(&deg[src[e]], 1.f);
}

__global__ void k_normk(const int* __restrict__ src, const int* __restrict__ dst,
                        const float* __restrict__ deg, float* __restrict__ nrm, int E) {
    int e = blockIdx.x * blockDim.x + threadIdx.x;
    if (e >= E) return;
    float a = deg[src[e]], b = deg[dst[e]];
    float da = a > 0.f ? rsqrtf(fmaxf(a, 1.f)) : 0.f;
    float db = b > 0.f ? rsqrtf(fmaxf(b, 1.f)) : 0.f;
    nrm[e] = -da * db;
}

// out[dst] += norm*x[src] per channel; 64 threads/block = channels
__global__ void k_prop(const float* __restrict__ x, const float* __restrict__ nrm,
                       const int* __restrict__ src, const int* __restrict__ dst,
                       float* __restrict__ out, int E) {
    int e = blockIdx.x;
    if (e >= E) return;
    int c = threadIdx.x;
    atomicAdd(&out[(size_t)dst[e] * 64 + c], nrm[e] * x[(size_t)src[e] * 64 + c]);
}

__global__ void k_axpby(const float* __restrict__ a, const float* __restrict__ b,
                        float* __restrict__ y, int n) {   // y = 2a - b
    int i = blockIdx.x * blockDim.x + threadIdx.x;
    if (i < n) y[i] = 2.f * a[i] - b[i];
}

// ---------------- final blend: out = Y(chlast) + rgb*sigmoid(G+bias) + rgb ----------------
__global__ void k_final(const float* __restrict__ Y, const float* __restrict__ G,
                        const float* __restrict__ bias, const float* __restrict__ rgb,
                        float* __restrict__ out, int total) {
    int i = blockIdx.x * blockDim.x + threadIdx.x;
    if (i >= total) return;
    int b = i / (cC * cHW);
    int c = (i / cHW) % cC;
    int hw = i % cHW;
    int m = b * cHW + hw;
    float wg = sigm(G[i] + bias[i & 63]);   // faithful raw reshape of [M,64] -> NCHW
    float r = rgb[i];
    out[i] = Y[(size_t)m * 64 + c] + r * wg + r;
}

// ---------------- host orchestration ----------------
extern "C" void kernel_launch(void* const* d_in, const int* in_sizes, int n_in,
                              void* d_out, int out_size, void* d_ws, size_t ws_size,
                              hipStream_t stream) {
    const float* rgb   = (const float*)d_in[0];
    const float* depth = (const float*)d_in[1];
    const float* wdp   = (const float*)d_in[2];
    auto P = [&](int i) -> const float* { return (const float*)d_in[i]; };
    // conv i (0..16): w=3+3i, g=4+3i, b=5+3i; order:
    // 0 cat1, 1 c32, 2 c33, 3 cat2, 4 out, 5 m1a, 6 m1b, 7 m3a, 8 m3b, 9 m3c,
    // 10 m5a, 11 m5b, 12 m5c, 13 s3, 14 s5, 15 s7, 16 gconv
    auto CW = [&](int i) { return P(3 + 3 * i); };
    auto CG = [&](int i) { return P(4 + 3 * i); };
    auto CBe = [&](int i) { return P(5 + 3 * i); };
    const float* cheb1w = P(54); const float* cheb2w = P(56);
    const float* cheb3w = P(58); const float* cheb3b = P(59);
    const float* gn1g = P(60); const float* gn1b = P(61);
    const float* gn2g = P(62); const float* gn2b = P(63);
    const int* esrc = (const int*)d_in[64];
    const int* edst = (const int*)d_in[65];
    int E = in_sizes[64];

    // ----- workspace arena -----
    float* ws = (float*)d_ws;
    const size_t MC = (size_t)cM * 64;
    float* FEAT   = ws;
    float* CCMOUT = FEAT + MC;
    float* BIGA   = CCMOUT + MC;          // [M,128] concat; later TXa/TXb
    float* TXa    = BIGA;
    float* TXb    = BIGA + MC;
    float* GOUT   = BIGA + 2 * MC;
    float* XCUR   = GOUT + MC;
    float* YOUT   = XCUR + MC;
    float* SM     = YOUT + MC;
    size_t so = 0;
    auto sal = [&](size_t n) { float* r = SM + so; so += n; return r; };
    float* X2 = sal((size_t)cM * 2);
    float* X2S = sal((size_t)cM * 2);
    float* WTMP = sal(cM);
    float* WSCM = sal(cM);
    float* SRAW = sal(cM);
    float* CAT2IN = sal((size_t)cM * 8);
    float* C2OUT = sal((size_t)cM * 4);
    float* DEG = sal(cM);
    float* NORM = sal(262144);
    float* SCALE = sal(64);
    float* SHIFT = sal(64);
    // packed-bf16 weight operands (u32 views; float slots are generous)
    u32* BPC1 = (u32*)sal(64 * 64);     // 128/2 x 64
    u32* BPG  = (u32*)sal(32 * 64);     // 64/2 x 64
    u32* BP32 = (u32*)sal(288 * 16);    // 576/2 x 16
    u32* BP33 = (u32*)sal(288 * 16);
    u32* BPC2 = (u32*)sal(48 * 16);     // 96/2 x 16
    u32* BPO  = (u32*)sal(32 * 64);     // 64/2 x 64
    u32* BCH1 = (u32*)sal(2048);        // 64/2 x 64
    u32* BCH2 = (u32*)sal(3 * 2048);
    u32* BCH3 = (u32*)sal(5 * 2048);
    float* PAVG = sal(512);
    float* PMAX = sal(512);
    float* W1 = sal(512);
    float* W3 = sal(512);
    float* CCMW = sal(512);
    float* AP = sal(12800);
    float* MPb = sal(12800);
    float* T1 = sal(1024);
    float* T2 = sal(1024);
    float* OA = sal(512);
    float* OB = sal(512);

    auto zero = [&](float* p, size_t n) {
        k_zero<<<(int)((n + 255) / 256), 256, 0, stream>>>(p, (int)n);
    };
    auto repack = [&](const float* w, u32* Bp, int O, int I, int KH, int KW, int Kp, int Np) {
        int tot = (Kp / 2) * Np;
        k_repack_pk<<<(tot + 255) / 256, 256, 0, stream>>>(w, Bp, O, I, KH, KW, Kp, Np);
    };
    auto packB = [&](const float* W, u32* Bp, int Kd, int N, int Kp, int Np) {
        int tot = (Kp / 2) * Np;
        k_pack_bmat<<<(tot + 255) / 256, 256, 0, stream>>>(W, Bp, Kd, N, Kp, Np);
    };
    // dense GEMMs: only two shapes exist (cat1: lda128/K128; GAT family: lda64/K64), Npad=64
    const int gw = (cM / 16) * 4;            // waves
    const int gb = (gw + 7) / 8;             // blocks of 256
    auto gemm128 = [&](const float* A, const u32* Bp, float* Cm, bool acc) {
        k_gemm<128, 64, 64, 128><<<gb, 256, 0, stream>>>(A, Bp, Cm, 4, gw, acc ? 1 : 0);
    };
    auto gemm64 = [&](const float* A, const u32* Bp, float* Cm, bool acc) {
        k_gemm<64, 64, 64, 64><<<gb, 256, 0, stream>>>(A, Bp, Cm, 4, gw, acc ? 1 : 0);
    };
    // conv3x3 instantiations
    const int cw1 = (cM / 16) * 1, cb1 = (cw1 + 7) / 8;   // Npad=16
    const int cw4 = (cM / 16) * 4, cb4 = (cw4 + 7) / 8;   // Npad=64
    auto conv_c3 = [&](const float* In, const u32* Bp, float* Cm) {   // 64ch -> 16pad
        k_conv3<64, 64, 576, 16, 16><<<cb1, 256, 0, stream>>>(In, Bp, Cm, 1, cw1);
    };
    auto conv_cat2 = [&](const float* In, const u32* Bp, float* Cm) { // 8ch -> 16pad
        k_conv3<8, 8, 96, 16, 16><<<cb1, 256, 0, stream>>>(In, Bp, Cm, 1, cw1);
    };
    auto conv_out = [&](const float* In, const u32* Bp, float* Cm) {  // 4ch -> 64
        k_conv3<4, 4, 64, 64, 64><<<cb4, 256, 0, stream>>>(In, Bp, Cm, 4, cw4);
    };
    auto bnstage = [&](const float* X, int ldx, int N, const float* g, const float* b,
                       float* Y, int ldy, int mode, const float* rows) {
        k_colstats<<<N, 256, 0, stream>>>(X, ldx, g, b, SCALE, SHIFT);
        int tot = cM * N;
        k_bn_act<<<(tot + 255) / 256, 256, 0, stream>>>(X, ldx, Y, ldy, N, SCALE, SHIFT, mode, rows);
    };
    auto run_mlp = [&](const float* in, int S,
                       const float* wA, const float* gA, const float* bA,
                       const float* wB, const float* gB, const float* bB,
                       const float* wC, const float* gC, const float* bC, float* out) {
        int S2 = S * S;
        k_tiny_conv1x1<<<1, 256, 0, stream>>>(in, wA, T1, cB, 64, 4, S2);
        k_tiny_bnrelu<<<1, 256, 0, stream>>>(T1, cB, 4, S2, gA, bA);
        const float* cur = T1;
        if (wB) {
            k_tiny_convk<<<1, 256, 0, stream>>>(T1, wB, T2, cB, 4, 4, S);
            k_tiny_bnrelu<<<1, 256, 0, stream>>>(T2, cB, 4, 1, gB, bB);
            cur = T2;
        }
        k_tiny_conv1x1<<<1, 256, 0, stream>>>(cur, wC, out, cB, 4, 64, 1);
        k_tiny_bnrelu<<<1, 256, 0, stream>>>(out, cB, 64, 1, gC, bC);
    };
    auto prop = [&](const float* x, float* out) {
        zero(out, MC);
        k_prop<<<E, 64, 0, stream>>>(x, NORM, esrc, edst, out, E);
    };

    // ----- weight prepack (packed bf16 B operands) -----
    repack(CW(0), BPC1, 64, 128, 1, 1, 128, 64);
    repack(CW(16), BPG, 64, 64, 1, 1, 64, 64);
    repack(CW(1), BP32, 4, 64, 3, 3, 576, 16);
    repack(CW(2), BP33, 4, 64, 3, 3, 576, 16);
    repack(CW(3), BPC2, 4, 8, 3, 3, 96, 16);
    repack(CW(4), BPO, 64, 4, 3, 3, 64, 64);
    packB(cheb1w, BCH1, 64, 64, 64, 64);
    for (int k = 0; k < 3; ++k) packB(cheb2w + (size_t)k * 4096, BCH2 + (size_t)k * 2048, 64, 64, 64, 64);
    for (int k = 0; k < 5; ++k) packB(cheb3w + (size_t)k * 4096, BCH3 + (size_t)k * 2048, 64, 64, 64, 64);

    const int totMC = cM * 64;
    // ----- feature = basic(concat(rgb,depth) @ cat1) * weight_d -----
    k_pack_nchw<<<(totMC + 255) / 256, 256, 0, stream>>>(rgb, BIGA, 128, 0);
    k_pack_nchw<<<(totMC + 255) / 256, 256, 0, stream>>>(depth, BIGA, 128, 64);
    gemm128(BIGA, BPC1, GOUT, false);
    bnstage(GOUT, 64, 64, CG(0), CBe(0), FEAT, 64, 0, wdp);

    // ----- CCM -----
    k_colpool<<<512, 256, 0, stream>>>(FEAT, PAVG, PMAX);
    run_mlp(PAVG, 1, CW(5), CG(5), CBe(5), nullptr, nullptr, nullptr, CW(6), CG(6), CBe(6), OA);
    run_mlp(PMAX, 1, CW(5), CG(5), CBe(5), nullptr, nullptr, nullptr, CW(6), CG(6), CBe(6), OB);
    k_addsig<<<2, 256, 0, stream>>>(OA, OB, W1, 512);
    k_blockpool<<<512 * 9, 256, 0, stream>>>(FEAT, W1, 3, AP, MPb);
    run_mlp(AP, 3, CW(7), CG(7), CBe(7), CW(8), CG(8), CBe(8), CW(9), CG(9), CBe(9), OA);
    run_mlp(MPb, 3, CW(7), CG(7), CBe(7), CW(8), CG(8), CBe(8), CW(9), CG(9), CBe(9), OB);
    k_addsig<<<2, 256, 0, stream>>>(OA, OB, W3, 512);
    k_blockpool<<<512 * 25, 256, 0, stream>>>(FEAT, W3, 5, AP, MPb);
    run_mlp(AP, 5, CW(10), CG(10), CBe(10), CW(11), CG(11), CBe(11), CW(12), CG(12), CBe(12), OA);
    run_mlp(MPb, 5, CW(10), CG(10), CBe(10), CW(11), CG(11), CBe(11), CW(12), CG(12), CBe(12), OB);
    k_addsig<<<2, 256, 0, stream>>>(OA, OB, CCMW, 512);
    k_scale_bc<<<(totMC + 255) / 256, 256, 0, stream>>>(FEAT, CCMW, CCMOUT);

    // ----- SCM -----
    k_rowreduce<<<(cM + 255) / 256, 256, 0, stream>>>(CCMOUT, X2);
    k_conv2to1<<<(cM + 255) / 256, 256, 0, stream>>>(X2, CW(13), 3, SRAW);
    bnstage(SRAW, 1, 1, CG(13), CBe(13), WTMP, 1, 1, nullptr);
    k_mul2<<<(cM * 2 + 255) / 256, 256, 0, stream>>>(X2, WTMP, X2S);
    k_conv2to1<<<(cM + 255) / 256, 256, 0, stream>>>(X2S, CW(14), 5, SRAW);
    bnstage(SRAW, 1, 1, CG(14), CBe(14), WTMP, 1, 1, nullptr);
    k_mul2<<<(cM * 2 + 255) / 256, 256, 0, stream>>>(X2, WTMP, X2S);
    k_conv2to1<<<(cM + 255) / 256, 256, 0, stream>>>(X2S, CW(15), 7, SRAW);
    bnstage(SRAW, 1, 1, CG(15), CBe(15), WSCM, 1, 0, nullptr);

    // ----- s1 / s2 / cat2 / out conv -----
    k_make_s12<<<(totMC + 255) / 256, 256, 0, stream>>>(CCMOUT, WSCM, XCUR, YOUT);
    conv_c3(XCUR, BP32, GOUT);
    bnstage(GOUT, 16, 4, CG(1), CBe(1), CAT2IN, 8, 0, nullptr);
    conv_c3(YOUT, BP33, GOUT);
    bnstage(GOUT, 16, 4, CG(2), CBe(2), CAT2IN + 4, 8, 0, nullptr);
    conv_cat2(CAT2IN, BPC2, GOUT);
    bnstage(GOUT, 16, 4, CG(3), CBe(3), C2OUT, 4, 0, nullptr);
    conv_out(C2OUT, BPO, GOUT);
    bnstage(GOUT, 64, 64, CG(4), CBe(4), YOUT, 64, 0, nullptr);

    // ----- GAT -----
    gemm64(FEAT, BPG, GOUT, false);
    bnstage(GOUT, 64, 64, CG(16), CBe(16), XCUR, 64, 0, nullptr);
    zero(DEG, cM);
    k_deg<<<(E + 255) / 256, 256, 0, stream>>>(esrc, DEG, E);
    k_normk<<<(E + 255) / 256, 256, 0, stream>>>(esrc, edst, DEG, NORM, E);
    // cheb1 (K=1)
    gemm64(XCUR, BCH1, GOUT, false);
    bnstage(GOUT, 64, 64, gn1g, gn1b, XCUR, 64, 0, nullptr);
    // cheb2 (K=3)
    gemm64(XCUR, BCH2, GOUT, false);
    prop(XCUR, TXa);
    gemm64(TXa, BCH2 + 2048, GOUT, true);
    prop(TXa, TXb);
    k_axpby<<<(totMC + 255) / 256, 256, 0, stream>>>(TXb, XCUR, TXb, totMC);
    gemm64(TXb, BCH2 + 2 * 2048, GOUT, true);
    bnstage(GOUT, 64, 64, gn2g, gn2b, XCUR, 64, 0, nullptr);
    // cheb3 (K=5): P0=XCUR, P1=TXa, P2=TXb
    gemm64(XCUR, BCH3, GOUT, false);
    prop(XCUR, TXa);
    gemm64(TXa, BCH3 + 2048, GOUT, true);
    prop(TXa, TXb);
    k_axpby<<<(totMC + 255) / 256, 256, 0, stream>>>(TXb, XCUR, TXb, totMC);
    gemm64(TXb, BCH3 + 2 * 2048, GOUT, true);
    prop(TXb, XCUR);
    k_axpby<<<(totMC + 255) / 256, 256, 0, stream>>>(XCUR, TXa, XCUR, totMC);
    gemm64(XCUR, BCH3 + 3 * 2048, GOUT, true);
    prop(XCUR, TXa);
    k_axpby<<<(totMC + 255) / 256, 256, 0, stream>>>(TXa, TXb, TXa, totMC);
    gemm64(TXa, BCH3 + 4 * 2048, GOUT, true);

    // ----- final blend -----
    int total = cB * cC * cHW;
    k_final<<<(total + 255) / 256, 256, 0, stream>>>(YOUT, GOUT, cheb3b, rgb,
                                                     (float*)d_out, total);
    (void)n_in; (void)out_size; (void)ws_size; (void)in_sizes;
}
